// MultiheadPerformerAttention_57698590654981
// MI455X (gfx1250) — matrix operand order repaired and checked by reading, not compiled, not run on hardware
//
#include <hip/hip_runtime.h>
#include <math.h>

// ---------------------------------------------------------------------------
// Types for CDNA5 WMMA (wave32, v_wmma_f32_16x16x32_bf16)
// ---------------------------------------------------------------------------
typedef __bf16 bf16;
typedef __attribute__((ext_vector_type(16))) __bf16 v16bf;
typedef __attribute__((ext_vector_type(8)))  __bf16 v8bf;
typedef __attribute__((ext_vector_type(8)))  float  v8f;

#define DEVI __device__ __forceinline__

// Problem constants (from setup_inputs)
#define BB  4
#define NN  4096
#define DD  1024
#define HH  16
#define JJ  64
#define DH  64
#define BH  (BB*HH)          // 64
#define MM  (BB*NN)          // 16384

#define CNORM       0.17677669529663687f   // 1024^-0.25
#define HALF_CN2    0.015625f              // 0.5 * 1024^-0.5
#define RATIO       0.125f                 // 64^-0.5
#define EPS_FEAT    1e-4f
#define EPS_DEN     0.01f

// ---------------------------------------------------------------------------
// WMMA fragment loaders (direct from global; operands are L2 resident)
// A (16x32, 16-bit): lane<16 -> row=lane, K in {0..7}U{16..23};
//                    lane>=16 -> row=lane-16, K in {8..15}U{24..31}
// B (32x16, 16-bit): lane -> col=lane&15, K same as A (strided rows)
// ---------------------------------------------------------------------------
DEVI v16bf load_frag_a(const bf16* __restrict__ p, int lda) {
  const int lane = threadIdx.x & 31;
  const int row  = lane & 15;
  const int koff = (lane & 16) ? 8 : 0;
  const bf16* q = p + (size_t)row * lda + koff;
  v8bf lo = *(const v8bf*)(q);
  v8bf hi = *(const v8bf*)(q + 16);
  v16bf f;
#pragma unroll
  for (int i = 0; i < 8; ++i) { f[i] = lo[i]; f[i + 8] = hi[i]; }
  return f;
}

DEVI v16bf load_frag_b(const bf16* __restrict__ p, int ldb) {
  const int lane = threadIdx.x & 31;
  const int koff = (lane & 16) ? 8 : 0;
  const bf16* q = p + (size_t)koff * ldb + (lane & 15);
  v16bf f;
#pragma unroll
  for (int i = 0; i < 8; ++i) { f[i] = q[(size_t)i * ldb]; f[i + 8] = q[(size_t)(16 + i) * ldb]; }
  return f;
}

DEVI v8f wmma_bf16(v16bf a, v16bf b, v8f c) {
  return __builtin_amdgcn_wmma_f32_16x16x32_bf16(
      false, a, false, b, (short)0, c, false, false);
}

// sortable-uint encoding of float for deterministic atomic max
DEVI unsigned enc_f32(float x) {
  unsigned u = __float_as_uint(x);
  return (u & 0x80000000u) ? ~u : (u | 0x80000000u);
}
DEVI float dec_f32(unsigned e) {
  unsigned u = (e & 0x80000000u) ? (e & 0x7FFFFFFFu) : ~e;
  return __uint_as_float(u);
}

// ---------------------------------------------------------------------------
// fp32 -> bf16 convert (grid-stride)
// ---------------------------------------------------------------------------
__global__ void k_f32_to_bf16(const float* __restrict__ in, bf16* __restrict__ out,
                              size_t n) {
  size_t i = (size_t)blockIdx.x * blockDim.x + threadIdx.x;
  size_t st = (size_t)gridDim.x * blockDim.x;
  for (; i < n; i += st) out[i] = (bf16)in[i];
}

// ---------------------------------------------------------------------------
// Batched LDS-tiled transpose + convert: in fp32 [R,C] -> out bf16 [C,R]
// block (32,8); grid (C/32, R/32, batch)
// ---------------------------------------------------------------------------
__global__ void k_transpose_conv(const float* __restrict__ in, bf16* __restrict__ out,
                                 int R, int C) {
  __shared__ float tile[32][33];
  const size_t base = (size_t)blockIdx.z * R * C;
  in  += base;
  out += base;
  const int r0 = blockIdx.y * 32, c0 = blockIdx.x * 32;
#pragma unroll
  for (int i = threadIdx.y; i < 32; i += 8)
    tile[i][threadIdx.x] = in[(size_t)(r0 + i) * C + c0 + threadIdx.x];
  __syncthreads();
#pragma unroll
  for (int i = threadIdx.y; i < 32; i += 8)
    out[(size_t)(c0 + i) * R + r0 + threadIdx.x] = (bf16)tile[threadIdx.x][i];
}

// ---------------------------------------------------------------------------
// Big NN GEMM: C[M,N] = A[M,K] * B[K,N] + bias[N]
// Block tile 128x128, 8 waves, each wave 32(M) x 64(N) = 2x4 WMMA frags.
// OUT_BF16: 1 -> write bf16, 0 -> write fp32
// grid (N/128, M/128), block 256
// ---------------------------------------------------------------------------
template <int OUT_BF16>
__global__ void k_gemm_nn_bias(const bf16* __restrict__ A, const bf16* __restrict__ B,
                               const float* __restrict__ bias, void* __restrict__ Cout,
                               int K, int lda, int ldb, int ldc) {
  const int wave = threadIdx.x >> 5;
  const int lane = threadIdx.x & 31;
  const int bm = blockIdx.y * 128 + (wave & 3) * 32;
  const int bn = blockIdx.x * 128 + (wave >> 2) * 64;

  v8f acc[2][4];
#pragma unroll
  for (int m = 0; m < 2; ++m)
#pragma unroll
    for (int f = 0; f < 4; ++f) acc[m][f] = (v8f)0.f;

  const bf16* Ap0 = A + (size_t)bm * lda;
  const bf16* Ap1 = A + (size_t)(bm + 16) * lda;
  const bf16* Bp  = B + bn;

  for (int k = 0; k < K; k += 32) {
    __builtin_prefetch(Ap0 + (size_t)0 * lda + k + 256, 0, 1);    // stream hint
    __builtin_prefetch(Bp + (size_t)(k + 256) * ldb, 0, 1);
    v16bf a0 = load_frag_a(Ap0 + k, lda);
    v16bf a1 = load_frag_a(Ap1 + k, lda);
#pragma unroll
    for (int f = 0; f < 4; ++f) {
      v16bf bfr = load_frag_b(Bp + (size_t)k * ldb + 16 * f, ldb);
      acc[0][f] = wmma_bf16(a0, bfr, acc[0][f]);
      acc[1][f] = wmma_bf16(a1, bfr, acc[1][f]);
    }
  }

  const int crow = (lane & 16) ? 8 : 0;
  const int ccol = lane & 15;
#pragma unroll
  for (int m = 0; m < 2; ++m)
#pragma unroll
    for (int f = 0; f < 4; ++f) {
      const int col = bn + 16 * f + ccol;
      const float bv = bias[col];
#pragma unroll
      for (int i = 0; i < 8; ++i) {
        const int row = bm + 16 * m + crow + i;
        const float v = acc[m][f][i] + bv;
        if (OUT_BF16)
          ((bf16*)Cout)[(size_t)row * ldc + col] = (bf16)v;
        else
          ((float*)Cout)[(size_t)row * ldc + col] = v;
      }
    }
}

// ---------------------------------------------------------------------------
// diag[bh][n] = 0.5*d^-0.5 * sum_d X[b,n,h*64+d]^2    (X is Q or K, bf16)
// grid (N/256, BH), block 256
// ---------------------------------------------------------------------------
__global__ void k_rowsq(const bf16* __restrict__ X, float* __restrict__ diag) {
  const int bh = blockIdx.y, b = bh >> 4, h = bh & 15;
  const int n = blockIdx.x * 256 + threadIdx.x;
  const bf16* r = X + ((size_t)b * NN + n) * DD + h * DH;
  float s = 0.f;
#pragma unroll
  for (int d = 0; d < DH; ++d) { float v = (float)r[d]; s += v * v; }
  diag[(size_t)bh * NN + n] = s * HALF_CN2;
}

// ---------------------------------------------------------------------------
// FAVOR+ feature kernel: dd = CNORM * (X_head @ projT[h])  via WMMA.
// MODE 0: query  -> rowmax stab, write q' [bh,n,j] bf16
// MODE 1: key A  -> per-head max via encoded atomicMax (deterministic)
// MODE 2: key B  -> use head stab, write k'^T [bh,j,n] bf16
// grid (N/128, BH), block 256; wave w owns rows n0+16w x all 64 j
// ---------------------------------------------------------------------------
template <int MODE>
__global__ void k_favor(const bf16* __restrict__ X, const bf16* __restrict__ projT,
                        const float* __restrict__ diag, unsigned* __restrict__ headmax,
                        bf16* __restrict__ outp) {
  const int bh = blockIdx.y, b = bh >> 4, h = bh & 15;
  const int wave = threadIdx.x >> 5;
  const int lane = threadIdx.x & 31;
  const int n0 = blockIdx.x * 128 + wave * 16;

  const bf16* Ap = X + ((size_t)b * NN + n0) * DD + h * DH;   // lda = DD
  const bf16* Bp = projT + (size_t)h * DH * JJ;               // [d][j], ldb = JJ

  v8f acc[4];
#pragma unroll
  for (int f = 0; f < 4; ++f) acc[f] = (v8f)0.f;

#pragma unroll
  for (int ks = 0; ks < 2; ++ks) {
    v16bf a = load_frag_a(Ap + ks * 32, DD);
#pragma unroll
    for (int f = 0; f < 4; ++f) {
      v16bf bfr = load_frag_b(Bp + (size_t)(ks * 32) * JJ + 16 * f, JJ);
      acc[f] = wmma_bf16(a, bfr, acc[f]);
    }
  }

  const int crow = (lane & 16) ? 8 : 0;
  const int ccol = lane & 15;

  float vals[4][8];
#pragma unroll
  for (int f = 0; f < 4; ++f)
#pragma unroll
    for (int i = 0; i < 8; ++i) vals[f][i] = acc[f][i] * CNORM;

  if (MODE == 1) {                    // per-head max of data_dash
    float m = -3.4e38f;
#pragma unroll
    for (int f = 0; f < 4; ++f)
#pragma unroll
      for (int i = 0; i < 8; ++i) m = fmaxf(m, vals[f][i]);
#pragma unroll
    for (int s = 1; s < 32; s <<= 1) m = fmaxf(m, __shfl_xor(m, s, 32));
    __shared__ float smax[8];
    if (lane == 0) smax[wave] = m;
    __syncthreads();
    if (threadIdx.x == 0) {
      float mm = smax[0];
#pragma unroll
      for (int w = 1; w < 8; ++w) mm = fmaxf(mm, smax[w]);
      atomicMax(&headmax[bh], enc_f32(mm));
    }
    return;
  }

  float stab_row[8];
  if (MODE == 0) {                    // per-row max over all 64 j
#pragma unroll
    for (int i = 0; i < 8; ++i) {
      float m = fmaxf(fmaxf(vals[0][i], vals[1][i]), fmaxf(vals[2][i], vals[3][i]));
#pragma unroll
      for (int s = 1; s < 16; s <<= 1) m = fmaxf(m, __shfl_xor(m, s, 32));
      stab_row[i] = m;
    }
  } else {                            // MODE 2: per-head stab
    const float st = dec_f32(headmax[bh]);
#pragma unroll
    for (int i = 0; i < 8; ++i) stab_row[i] = st;
  }

#pragma unroll
  for (int i = 0; i < 8; ++i) {
    const int n = n0 + crow + i;
    const float dg = diag[(size_t)bh * NN + n];
#pragma unroll
    for (int f = 0; f < 4; ++f) {
      const int j = 16 * f + ccol;
      const float fp = RATIO * __expf(vals[f][i] - dg - stab_row[i]) + EPS_FEAT;
      if (MODE == 0)                  // q' [bh,n,j]
        outp[((size_t)bh * NN + n) * JJ + j] = (bf16)fp;
      else                            // k'^T [bh,j,n]
        outp[((size_t)bh * JJ + j) * NN + n] = (bf16)fp;
    }
  }
}

// ---------------------------------------------------------------------------
// ksum[bh*64+j] = sum_n k'^T[bh,j,n]   -- grid BH*JJ, block 256
// ---------------------------------------------------------------------------
__global__ void k_ksum(const bf16* __restrict__ kT, float* __restrict__ ksum) {
  const bf16* row = kT + (size_t)blockIdx.x * NN;
  float s = 0.f;
  for (int i = threadIdx.x; i < NN; i += 256) s += (float)row[i];
  __shared__ float red[256];
  red[threadIdx.x] = s;
  __syncthreads();
#pragma unroll
  for (int st = 128; st > 0; st >>= 1) {
    if (threadIdx.x < st) red[threadIdx.x] += red[threadIdx.x + st];
    __syncthreads();
  }
  if (threadIdx.x == 0) ksum[blockIdx.x] = red[0];
}

// ---------------------------------------------------------------------------
// norm[bh,n] = q'[bh,n,:] . ksum[bh,:]   -- grid (N/256, BH), block 256
// ---------------------------------------------------------------------------
__global__ void k_normalizer(const bf16* __restrict__ qp, const float* __restrict__ ksum,
                             float* __restrict__ norm) {
  const int bh = blockIdx.y;
  __shared__ float ks[JJ];
  if (threadIdx.x < JJ) ks[threadIdx.x] = ksum[bh * JJ + threadIdx.x];
  __syncthreads();
  const int n = blockIdx.x * 256 + threadIdx.x;
  const bf16* r = qp + ((size_t)bh * NN + n) * JJ;
  float s = 0.f;
#pragma unroll
  for (int j = 0; j < JJ; ++j) s += (float)r[j] * ks[j];
  norm[(size_t)bh * NN + n] = s;
}

// ---------------------------------------------------------------------------
// kv[bh,j,d] = sum_n k'^T[bh,j,n] * V[b,n,h*64+d]   (64x64x4096 NN GEMM)
// grid BH, block 256; wave w: rows (w&3)*16, cols (w>>2)*32
// ---------------------------------------------------------------------------
__global__ void k_gemm_kv(const bf16* __restrict__ kT, const bf16* __restrict__ V,
                          bf16* __restrict__ kv) {
  const int bh = blockIdx.x, b = bh >> 4, h = bh & 15;
  const int wave = threadIdx.x >> 5;
  const int lane = threadIdx.x & 31;
  const int mr = (wave & 3) * 16;
  const int nc = (wave >> 2) * 32;

  const bf16* Ap = kT + ((size_t)bh * JJ + mr) * NN;             // lda = NN
  const bf16* Bp = V + (size_t)b * NN * DD + h * DH + nc;        // ldb = DD

  v8f acc[2];
  acc[0] = (v8f)0.f; acc[1] = (v8f)0.f;

  for (int k = 0; k < NN; k += 32) {
    v16bf a = load_frag_a(Ap + k, NN);
    v16bf b0 = load_frag_b(Bp + (size_t)k * DD, DD);
    v16bf b1 = load_frag_b(Bp + (size_t)k * DD + 16, DD);
    acc[0] = wmma_bf16(a, b0, acc[0]);
    acc[1] = wmma_bf16(a, b1, acc[1]);
  }

  const int crow = (lane & 16) ? 8 : 0;
  const int ccol = lane & 15;
#pragma unroll
  for (int f = 0; f < 2; ++f)
#pragma unroll
    for (int i = 0; i < 8; ++i) {
      const int j = mr + crow + i;
      const int d = nc + 16 * f + ccol;
      kv[((size_t)bh * JJ + j) * DH + d] = (bf16)acc[f][i];
    }
}

// ---------------------------------------------------------------------------
// attn[b,n,h*64+d] = (q'[bh,n,:] @ kv[bh,:,d]) / max(norm[bh,n], EPS_DEN)
// grid (N/128, BH), block 256; wave w: rows n0+16w x 64 cols
// ---------------------------------------------------------------------------
__global__ void k_gemm_qkv(const bf16* __restrict__ qp, const bf16* __restrict__ kv,
                           const float* __restrict__ norm, bf16* __restrict__ attn) {
  const int bh = blockIdx.y, b = bh >> 4, h = bh & 15;
  const int wave = threadIdx.x >> 5;
  const int lane = threadIdx.x & 31;
  const int n0 = blockIdx.x * 128 + wave * 16;

  const bf16* Ap = qp + ((size_t)bh * NN + n0) * JJ;   // lda = JJ
  const bf16* Bp = kv + (size_t)bh * JJ * DH;          // ldb = DH

  v8f acc[4];
#pragma unroll
  for (int f = 0; f < 4; ++f) acc[f] = (v8f)0.f;

#pragma unroll
  for (int ks = 0; ks < 2; ++ks) {
    v16bf a = load_frag_a(Ap + ks * 32, JJ);
#pragma unroll
    for (int f = 0; f < 4; ++f) {
      v16bf bfr = load_frag_b(Bp + (size_t)(ks * 32) * DH + 16 * f, DH);
      acc[f] = wmma_bf16(a, bfr, acc[f]);
    }
  }

  const int crow = (lane & 16) ? 8 : 0;
  const int ccol = lane & 15;
#pragma unroll
  for (int i = 0; i < 8; ++i) {
    const int n = n0 + crow + i;
    const float den = fmaxf(norm[(size_t)bh * NN + n], EPS_DEN);
    const float rden = 1.0f / den;
#pragma unroll
    for (int f = 0; f < 4; ++f) {
      const int d = 16 * f + ccol;
      attn[((size_t)b * NN + n) * DD + h * DH + d] = (bf16)(acc[f][i] * rden);
    }
  }
}

// ---------------------------------------------------------------------------
// Host launcher
// ---------------------------------------------------------------------------
extern "C" void kernel_launch(void* const* d_in, const int* in_sizes, int n_in,
                              void* d_out, int out_size, void* d_ws, size_t ws_size,
                              hipStream_t stream) {
  const float* query = (const float*)d_in[0];
  const float* Wq = (const float*)d_in[1];
  const float* bq = (const float*)d_in[2];
  const float* Wk = (const float*)d_in[3];
  const float* bk = (const float*)d_in[4];
  const float* Wv = (const float*)d_in[5];
  const float* bv = (const float*)d_in[6];
  const float* Wo = (const float*)d_in[7];
  const float* bo = (const float*)d_in[8];
  const float* proj = (const float*)d_in[9];

  char* p = (char*)d_ws;
  auto carve = [&](size_t bytes) -> char* {
    char* r = p;
    p += (bytes + 255) & ~(size_t)255;
    return r;
  };

  bf16* xb    = (bf16*)carve((size_t)MM * DD * 2);       // 32 MB
  bf16* WqT   = (bf16*)carve((size_t)DD * DD * 2);       // 2 MB each
  bf16* WkT   = (bf16*)carve((size_t)DD * DD * 2);
  bf16* WvT   = (bf16*)carve((size_t)DD * DD * 2);
  bf16* WoT   = (bf16*)carve((size_t)DD * DD * 2);
  bf16* projT = (bf16*)carve((size_t)HH * DH * JJ * 2);  // 128 KB
  bf16* Qb    = (bf16*)carve((size_t)MM * DD * 2);       // 32 MB each
  bf16* Kb    = (bf16*)carve((size_t)MM * DD * 2);
  bf16* Vb    = (bf16*)carve((size_t)MM * DD * 2);
  float* diag_q = (float*)carve((size_t)BH * NN * 4);    // 1 MB each
  float* diag_k = (float*)carve((size_t)BH * NN * 4);
  unsigned* headmax = (unsigned*)carve(256);
  bf16* qprime = (bf16*)carve((size_t)BH * NN * JJ * 2); // 32 MB
  bf16* kT     = (bf16*)carve((size_t)BH * JJ * NN * 2); // 32 MB
  float* ksum  = (float*)carve((size_t)BH * JJ * 4);
  bf16* kv     = (bf16*)carve((size_t)BH * JJ * DH * 2);
  float* norm  = (float*)carve((size_t)BH * NN * 4);     // 1 MB
  bf16* attn   = (bf16*)carve((size_t)MM * DD * 2);      // 32 MB
  (void)ws_size;

  // 1) convert input activations; transpose+convert weights and proj
  k_f32_to_bf16<<<4096, 256, 0, stream>>>(query, xb, (size_t)MM * DD);
  {
    dim3 blk(32, 8);
    dim3 gw(DD / 32, DD / 32, 1);
    k_transpose_conv<<<gw, blk, 0, stream>>>(Wq, WqT, DD, DD);
    k_transpose_conv<<<gw, blk, 0, stream>>>(Wk, WkT, DD, DD);
    k_transpose_conv<<<gw, blk, 0, stream>>>(Wv, WvT, DD, DD);
    k_transpose_conv<<<gw, blk, 0, stream>>>(Wo, WoT, DD, DD);
    dim3 gp(JJ / 32, DH / 32, HH);   // proj[h]: [J,dh] -> projT[h]: [dh,J]
    k_transpose_conv<<<gp, blk, 0, stream>>>(proj, projT, JJ, DH);
  }

  // 2) Q/K/V projections (bf16 out)
  {
    dim3 grid(DD / 128, MM / 128);
    k_gemm_nn_bias<1><<<grid, 256, 0, stream>>>(xb, WqT, bq, Qb, DD, DD, DD, DD);
    k_gemm_nn_bias<1><<<grid, 256, 0, stream>>>(xb, WkT, bk, Kb, DD, DD, DD, DD);
    k_gemm_nn_bias<1><<<grid, 256, 0, stream>>>(xb, WvT, bv, Vb, DD, DD, DD, DD);
  }

  // 3) diag terms
  {
    dim3 grid(NN / 256, BH);
    k_rowsq<<<grid, 256, 0, stream>>>(Qb, diag_q);
    k_rowsq<<<grid, 256, 0, stream>>>(Kb, diag_k);
  }

  // 4) FAVOR+ features
  hipMemsetAsync(headmax, 0, 256, stream);
  {
    dim3 grid(NN / 128, BH);
    k_favor<1><<<grid, 256, 0, stream>>>(Kb, projT, diag_k, headmax, kT);      // head max
    k_favor<0><<<grid, 256, 0, stream>>>(Qb, projT, diag_q, headmax, qprime);  // q'
    k_favor<2><<<grid, 256, 0, stream>>>(Kb, projT, diag_k, headmax, kT);      // k'^T
  }

  // 5) normalization pieces
  k_ksum<<<BH * JJ, 256, 0, stream>>>(kT, ksum);
  {
    dim3 grid(NN / 256, BH);
    k_normalizer<<<grid, 256, 0, stream>>>(qprime, ksum, norm);
  }

  // 6) linear attention contractions
  k_gemm_kv<<<BH, 256, 0, stream>>>(kT, Vb, kv);
  {
    dim3 grid(NN / 128, BH);
    k_gemm_qkv<<<grid, 256, 0, stream>>>(qprime, kv, norm, attn);
  }

  // 7) output projection (fp32 out)
  {
    dim3 grid(DD / 128, MM / 128);
    k_gemm_nn_bias<0><<<grid, 256, 0, stream>>>(attn, WoT, bo, d_out, DD, DD, DD, DD);
  }
}
